// ConcatCritic_3410204033673
// MI455X (gfx1250) — matrix operand order repaired and checked by reading, not compiled, not run on hardware
//
#include <hip/hip_runtime.h>
#include <hip/hip_bf16.h>

// ConcatCritic on MI455X (gfx1250): all-pairs 4-layer MLP.
// Layer1 factored into Ax[i] + Ay[j]; layers 2/3 run as bf16 WMMA
// (v_wmma_f32_16x16x32_bf16) with f32 accumulate; layer4 is a 256-dot.
// Register-blocked: each wave computes 32 rows x 128 cols, so every
// B (weight) fragment loaded from L2 feeds two WMMAs.

#define BN   512
#define DIMX 128
#define HIDN 256

typedef __attribute__((ext_vector_type(16))) __bf16 v16bf;
typedef __attribute__((ext_vector_type(8)))  float  v8f;

static __device__ __forceinline__ unsigned short f2bf(float f) {
  unsigned u = __float_as_uint(f);
  u += 0x7FFFu + ((u >> 16) & 1u);       // round-to-nearest-even
  return (unsigned short)(u >> 16);
}
static __device__ __forceinline__ float bf2f(unsigned short h) {
  return __uint_as_float(((unsigned)h) << 16);
}

// ---------------------------------------------------------------------------
// Prep: Ax = x @ W1[:128,:], Ay = y @ W1[128:,:] + b1 (f32),
//       W2t/W3t = bf16 transposed weights (Wt[n][k] = W[k][n]).
// ---------------------------------------------------------------------------
__global__ void concat_critic_prep(const float* __restrict__ x,
                                   const float* __restrict__ y,
                                   const float* __restrict__ W1,
                                   const float* __restrict__ b1,
                                   const float* __restrict__ W2,
                                   const float* __restrict__ W3,
                                   float* __restrict__ Ax,
                                   float* __restrict__ Ay,
                                   unsigned short* __restrict__ W2t,
                                   unsigned short* __restrict__ W3t) {
  const int NA = BN * HIDN;          // 131072
  const int NW = HIDN * HIDN;        // 65536
  int t = blockIdx.x * blockDim.x + threadIdx.x;
  if (t < NA) {
    int i = t >> 8, c = t & 255;
    float s = 0.f;
    const float* xr = x + i * DIMX;
#pragma unroll 4
    for (int k = 0; k < DIMX; ++k) s = fmaf(xr[k], W1[k * HIDN + c], s);
    Ax[t] = s;
  } else if (t < 2 * NA) {
    int u = t - NA;
    int j = u >> 8, c = u & 255;
    float s = b1[c];
    const float* yr = y + j * DIMX;
#pragma unroll 4
    for (int k = 0; k < DIMX; ++k) s = fmaf(yr[k], W1[(DIMX + k) * HIDN + c], s);
    Ay[u] = s;
  } else if (t < 2 * NA + NW) {
    int u = t - 2 * NA;
    int n = u >> 8, k = u & 255;
    W2t[u] = f2bf(W2[k * HIDN + n]);
  } else if (t < 2 * NA + 2 * NW) {
    int u = t - 2 * NA - NW;
    int n = u >> 8, k = u & 255;
    W3t[u] = f2bf(W3[k * HIDN + n]);
  }
}

// ---------------------------------------------------------------------------
// Main: each block = 8 waves = 128 pair rows (fixed i, 128 consecutive j).
// Each wave: 32 rows x 128 cols; B fragments reused across two A tiles.
// LDS double buffer of bf16 activations; layers 2/3 via WMMA bf16.
// ---------------------------------------------------------------------------
__global__ __launch_bounds__(256) void concat_critic_main(
    const float* __restrict__ Ax, const float* __restrict__ Ay,
    const unsigned short* __restrict__ W2t,
    const unsigned short* __restrict__ W3t,
    const float* __restrict__ b2, const float* __restrict__ b3,
    const float* __restrict__ W4, const float* __restrict__ b4,
    float* __restrict__ out) {
  __shared__ unsigned short hA[128 * HIDN];   // 64 KB
  __shared__ unsigned short hB[128 * HIDN];   // 64 KB

  const int tid   = threadIdx.x;
  const int bp    = blockIdx.x;        // 0..2047
  const int i     = bp >> 2;           // output row (x index)
  const int jbase = (bp & 3) << 7;     // 128 output cols (y indices)

  // ---- layer 1: h1[r][c] = relu(Ax[i][c] + Ay[jbase+r][c]) -> bf16 LDS
  for (int e = tid; e < 128 * HIDN; e += 256) {
    int r = e >> 8, c = e & 255;
    float v = Ax[i * HIDN + c] + Ay[(jbase + r) * HIDN + c];
    hA[e] = f2bf(fmaxf(v, 0.f));
  }
  __syncthreads();

  const int lane  = tid & 31;
  const int wv    = tid >> 5;          // wave 0..7
  const int mbase = (wv >> 1) << 5;    // 32-row group (4 groups)
  const int nhalf = (wv & 1) << 7;     // column half: 0 or 128
  const int lh    = lane >> 4;         // lane half (0/1)
  const int lm    = lane & 15;

  // ---- layers 2 and 3 (identical): hin -> WMMA -> bias+relu -> hout
#pragma unroll
  for (int layer = 0; layer < 2; ++layer) {
    const unsigned short* hin  = layer ? hB : hA;
    unsigned short*       hout = layer ? hA : hB;
    const unsigned short* Wt   = layer ? W3t : W2t;
    const float*          bias = layer ? b3 : b2;

    v8f acc[2][8];
#pragma unroll
    for (int u = 0; u < 2; ++u)
#pragma unroll
      for (int t = 0; t < 8; ++t)
        acc[u][t] = (v8f){0.f, 0.f, 0.f, 0.f, 0.f, 0.f, 0.f, 0.f};

#pragma unroll
    for (int s = 0; s < 8; ++s) {      // K steps of 32
      // Two A fragments (16x32 bf16 each): rows [mbase,mbase+16) and
      // [mbase+16,mbase+32). Lane half h holds contiguous K=[16h,16h+16).
      union { v16bf v; uint4 q[2]; } afr0, afr1;
      const unsigned short* arow0 =
          hin + (mbase + lm) * HIDN + (s << 5) + (lh << 4);
      const unsigned short* arow1 = arow0 + 16 * HIDN;
      afr0.q[0] = *(const uint4*)(arow0);
      afr0.q[1] = *(const uint4*)(arow0 + 8);
      afr1.q[0] = *(const uint4*)(arow1);
      afr1.q[1] = *(const uint4*)(arow1 + 8);

#pragma unroll
      for (int t = 0; t < 8; ++t) {
        // B fragment (32x16 bf16): lane = column, contiguous K=16h..16h+15.
        // Loaded once, consumed by two WMMAs (2x reuse vs L2).
        const unsigned short* bcol =
            Wt + (nhalf + (t << 4) + lm) * HIDN + (s << 5) + (lh << 4);
        v16bf bfr = *(const v16bf*)(bcol);
        acc[0][t] = __builtin_amdgcn_wmma_f32_16x16x32_bf16(
            false, afr0.v, false, bfr, (short)0, acc[0][t], false, false);
        acc[1][t] = __builtin_amdgcn_wmma_f32_16x16x32_bf16(
            false, afr1.v, false, bfr, (short)0, acc[1][t], false, false);
      }
    }

    // bias + relu + f32->bf16, spill C/D fragments (row = r + 8*lh, col = lm)
#pragma unroll
    for (int t = 0; t < 8; ++t) {
      int n = nhalf + (t << 4) + lm;
      float bv = bias[n];
#pragma unroll
      for (int u = 0; u < 2; ++u) {
#pragma unroll
        for (int r = 0; r < 8; ++r) {
          int m = mbase + (u << 4) + r + (lh << 3);
          hout[m * HIDN + n] = f2bf(fmaxf(acc[u][t][r] + bv, 0.f));
        }
      }
    }
    __syncthreads();
  }

  // ---- layer 4 + softplus: h3 (bf16) lives in hA; one thread per row
  if (tid < 128) {
    int r = tid;
    float s = b4[0];
    const unsigned short* hr = hA + r * HIDN;
#pragma unroll 4
    for (int n = 0; n < HIDN; ++n) s = fmaf(bf2f(hr[n]), W4[n], s);
    float o = (s > 20.f) ? s : log1pf(__expf(s));
    out[i * BN + (jbase + r)] = o;     // out[i][j] = f(x[i], y[j])
  }
}

// ---------------------------------------------------------------------------
extern "C" void kernel_launch(void* const* d_in, const int* in_sizes, int n_in,
                              void* d_out, int out_size, void* d_ws, size_t ws_size,
                              hipStream_t stream) {
  const float* x  = (const float*)d_in[0];
  const float* y  = (const float*)d_in[1];
  const float* W1 = (const float*)d_in[2];
  const float* b1 = (const float*)d_in[3];
  const float* W2 = (const float*)d_in[4];
  const float* b2 = (const float*)d_in[5];
  const float* W3 = (const float*)d_in[6];
  const float* b3 = (const float*)d_in[7];
  const float* W4 = (const float*)d_in[8];
  const float* b4 = (const float*)d_in[9];
  float* out = (float*)d_out;

  char* ws = (char*)d_ws;
  float*          Ax  = (float*)(ws);
  float*          Ay  = (float*)(ws + (size_t)BN * HIDN * 4);
  unsigned short* W2t = (unsigned short*)(ws + (size_t)2 * BN * HIDN * 4);
  unsigned short* W3t = (unsigned short*)(ws + (size_t)2 * BN * HIDN * 4
                                             + (size_t)HIDN * HIDN * 2);

  // prep work items: 2*512*256 + 2*256*256 = 393216 -> 1536 blocks of 256
  concat_critic_prep<<<1536, 256, 0, stream>>>(x, y, W1, b1, W2, W3,
                                               Ax, Ay, W2t, W3t);
  // 512*512 pairs / 128 rows per block = 2048 blocks
  concat_critic_main<<<2048, 256, 0, stream>>>(Ax, Ay, W2t, W3t,
                                               b2, b3, W4, b4, out);
}
